// Decoder_7421703487621
// MI455X (gfx1250) — matrix operand order repaired and checked by reading, not compiled, not run on hardware
//
#include <hip/hip_runtime.h>
#include <cstddef>
#include <cstdint>

// ---------------------------------------------------------------------------
// MI455X / gfx1250 attention-LSTM decoder.
// WMMA f32_16x16x32_f16 for all GEMMs, 4 N-tiles per wave (A-fragment reuse);
// ctx cached in f16 (64 MB, fits the 192 MB L2).
// ---------------------------------------------------------------------------

typedef __attribute__((ext_vector_type(16))) _Float16 v16h;
typedef __attribute__((ext_vector_type(8)))  float    v8f;

#define B_  32
#define S_  2048
#define T_  32
#define H_  512
#define E_  256
#define NL_ 2000

__device__ __forceinline__ float sigmoidf_(float x) { return 1.0f / (1.0f + __expf(-x)); }

// ---------------------------------------------------------------------------
// A-fragment (16x32 f16, M x K) loaded from f32 row-major matrix.
// Layout: lanes 0-15 hold M=0..15 with K = 0..15,
//         lanes 16-31 hold M=0..15 with K = 16..31 (matches swizzled B).
// ---------------------------------------------------------------------------
__device__ __forceinline__ v16h load_a_frag(const float* __restrict__ A, int lda,
                                            int m0, int k0, int lane)
{
    int m = m0 + (lane & 15);
    const float* row = A + (size_t)m * lda + k0 + ((lane >> 4) << 4);
    float4 a0 = *(const float4*)(row + 0);
    float4 a1 = *(const float4*)(row + 4);
    float4 b0 = *(const float4*)(row + 8);
    float4 b1 = *(const float4*)(row + 12);
    v16h v;
    v[0]=(_Float16)a0.x;  v[1]=(_Float16)a0.y;  v[2]=(_Float16)a0.z;  v[3]=(_Float16)a0.w;
    v[4]=(_Float16)a1.x;  v[5]=(_Float16)a1.y;  v[6]=(_Float16)a1.z;  v[7]=(_Float16)a1.w;
    v[8]=(_Float16)b0.x;  v[9]=(_Float16)b0.y;  v[10]=(_Float16)b0.z; v[11]=(_Float16)b0.w;
    v[12]=(_Float16)b1.x; v[13]=(_Float16)b1.y; v[14]=(_Float16)b1.z; v[15]=(_Float16)b1.w;
    return v;
}

// ---------------------------------------------------------------------------
// Pre-swizzle W (N x K, row-major f32; used as B = W^T) into fragment-major
// f16: tile (nt,kt) is 512 f16; lane L owns 16 contiguous f16 =
// B[k = (L>>4)*16 + e][n = nt*16 + (L&15)].  One 32B load/lane at GEMM time.
// ---------------------------------------------------------------------------
__global__ void swizzle_b(const float* __restrict__ W, _Float16* __restrict__ Bs,
                          int N, int K, int Ntiles, int Ktiles)
{
    size_t idx = (size_t)blockIdx.x * blockDim.x + threadIdx.x;
    size_t total = (size_t)Ntiles * Ktiles * 512;
    if (idx >= total) return;
    int e    = idx & 15;
    int lane = (idx >> 4) & 31;
    size_t tile = idx >> 9;
    int kt = (int)(tile % Ktiles);
    int nt = (int)(tile / Ktiles);
    int k = kt * 32 + ((lane >> 4) << 4) + e;
    int n = nt * 16 + (lane & 15);
    float v = (n < N && k < K) ? W[(size_t)n * K + k] : 0.0f;
    Bs[idx] = (_Float16)v;
}

// Combined gates weight: K = 768 = [W_ih (k<256) | W_hh (k>=256)], N = 2048.
__global__ void swizzle_gates(const float* __restrict__ Wih,
                              const float* __restrict__ Whh,
                              _Float16* __restrict__ Bs)
{
    const int Ktiles = 24, Ntiles = 128;
    size_t idx = (size_t)blockIdx.x * blockDim.x + threadIdx.x;
    size_t total = (size_t)Ntiles * Ktiles * 512;
    if (idx >= total) return;
    int e    = idx & 15;
    int lane = (idx >> 4) & 31;
    size_t tile = idx >> 9;
    int kt = (int)(tile % Ktiles);
    int nt = (int)(tile / Ktiles);
    int k = kt * 32 + ((lane >> 4) << 4) + e;
    int n = nt * 16 + (lane & 15);
    float v = (k < E_) ? Wih[(size_t)n * E_ + k]
                       : Whh[(size_t)n * H_ + (k - E_)];
    Bs[idx] = (_Float16)v;
}

// ---------------------------------------------------------------------------
// 4-wide WMMA GEMM: each wave computes a 16(M) x 64(N) strip = 4 WMMA tiles,
// reusing one A fragment per K-tile (4x less A traffic, 4x WMMA ILP).
// C[M x Nreal] = A[M x K](f32) @ Bs + bias(+bias2), optional tanh, optional C2.
// ---------------------------------------------------------------------------
__global__ void gemm_wmma4(const float* __restrict__ A, int lda,
                           const _Float16* __restrict__ Bs, int Ktiles,
                           const float* __restrict__ bias,
                           const float* __restrict__ bias2,
                           float* __restrict__ C, int ldc,
                           float* __restrict__ C2, int ldc2,
                           int Mtiles, int Ngroups, int Nreal, int act)
{
    int wid  = (int)((blockIdx.x * blockDim.x + threadIdx.x) >> 5);
    int lane = threadIdx.x & 31;
    if (wid >= Mtiles * Ngroups) return;          // wave-uniform exit, EXEC all-1
    int ng = wid % Ngroups;
    int mt = wid / Ngroups;

    const _Float16* Bt = Bs + (size_t)(ng * 4) * Ktiles * 512;
    size_t tstride = (size_t)Ktiles * 512;
    v8f acc0 = {}, acc1 = {}, acc2 = {}, acc3 = {};
    for (int kt = 0; kt < Ktiles; ++kt) {
        v16h a  = load_a_frag(A, lda, mt * 16, kt * 32, lane);
        const _Float16* bp = Bt + (size_t)kt * 512 + (size_t)lane * 16;
        v16h b0 = *(const v16h*)(bp);
        v16h b1 = *(const v16h*)(bp + tstride);
        v16h b2 = *(const v16h*)(bp + 2 * tstride);
        v16h b3 = *(const v16h*)(bp + 3 * tstride);
        acc0 = __builtin_amdgcn_wmma_f32_16x16x32_f16(false, a, false, b0, (short)0, acc0, false, false);
        acc1 = __builtin_amdgcn_wmma_f32_16x16x32_f16(false, a, false, b1, (short)0, acc1, false, false);
        acc2 = __builtin_amdgcn_wmma_f32_16x16x32_f16(false, a, false, b2, (short)0, acc2, false, false);
        acc3 = __builtin_amdgcn_wmma_f32_16x16x32_f16(false, a, false, b3, (short)0, acc3, false, false);
    }

    int mbase = mt * 16 + ((lane >> 4) << 3);
    v8f accs[4] = {acc0, acc1, acc2, acc3};
#pragma unroll
    for (int i = 0; i < 4; ++i) {
        int n = (ng * 4 + i) * 16 + (lane & 15);
        if (n >= Nreal) continue;
        float bv = 0.0f;
        if (bias)  bv += bias[n];
        if (bias2) bv += bias2[n];
#pragma unroll
        for (int r = 0; r < 8; ++r) {
            float v = accs[i][r] + bv;
            if (act) v = tanhf(v);
            int m = mbase + r;
            C[(size_t)m * ldc + n] = v;
            if (C2) C2[(size_t)m * ldc2 + n] = v;
        }
    }
}

// ctx GEMM (4-wide): (B*S x 512) = context(f32) @ W_ctx^T + b_ctx, f16 output.
__global__ void gemm_wmma_ctx4(const float* __restrict__ A,
                               const _Float16* __restrict__ Bs,
                               const float* __restrict__ bias,
                               _Float16* __restrict__ C)
{
    const int Ktiles = 16, Ngroups = 8;           // Ntiles = 32
    int wid  = (int)((blockIdx.x * blockDim.x + threadIdx.x) >> 5);
    int lane = threadIdx.x & 31;
    int ng = wid & (Ngroups - 1);
    int mt = wid >> 3;                            // 0..4095

    const _Float16* Bt = Bs + (size_t)(ng * 4) * Ktiles * 512;
    const size_t tstride = (size_t)Ktiles * 512;
    v8f acc0 = {}, acc1 = {}, acc2 = {}, acc3 = {};
#pragma unroll 2
    for (int kt = 0; kt < Ktiles; ++kt) {
        v16h a  = load_a_frag(A, H_, mt * 16, kt * 32, lane);
        const _Float16* bp = Bt + (size_t)kt * 512 + (size_t)lane * 16;
        v16h b0 = *(const v16h*)(bp);
        v16h b1 = *(const v16h*)(bp + tstride);
        v16h b2 = *(const v16h*)(bp + 2 * tstride);
        v16h b3 = *(const v16h*)(bp + 3 * tstride);
        acc0 = __builtin_amdgcn_wmma_f32_16x16x32_f16(false, a, false, b0, (short)0, acc0, false, false);
        acc1 = __builtin_amdgcn_wmma_f32_16x16x32_f16(false, a, false, b1, (short)0, acc1, false, false);
        acc2 = __builtin_amdgcn_wmma_f32_16x16x32_f16(false, a, false, b2, (short)0, acc2, false, false);
        acc3 = __builtin_amdgcn_wmma_f32_16x16x32_f16(false, a, false, b3, (short)0, acc3, false, false);
    }
    int mbase = mt * 16 + ((lane >> 4) << 3);
    v8f accs[4] = {acc0, acc1, acc2, acc3};
#pragma unroll
    for (int i = 0; i < 4; ++i) {
        int n = (ng * 4 + i) * 16 + (lane & 15);
        float bv = bias[n];
#pragma unroll
        for (int r = 0; r < 8; ++r)
            C[(size_t)(mbase + r) * H_ + n] = (_Float16)(accs[i][r] + bv);
    }
}

// ---------------------------------------------------------------------------
// Per-step elementwise / attention kernels
// ---------------------------------------------------------------------------

// xh[b,0:256]=x_t, xh[b,256:768]=h_{t-1}; zout[b,1024:1280]=x_t
__global__ void build_x(const float* __restrict__ dec_in,
                        const int*   __restrict__ cur_labels,
                        const float* __restrict__ label_emb,
                        const float* __restrict__ h,
                        float* __restrict__ xh, float* __restrict__ zout, int t)
{
    int idx = blockIdx.x * blockDim.x + threadIdx.x;
    if (idx >= B_ * 768) return;
    int b = idx / 768, j = idx % 768;
    if (j < E_) {
        float x = (t == 0) ? dec_in[b * E_ + j]
                           : label_emb[(size_t)cur_labels[b * T_ + (t - 1)] * E_ + j];
        xh[idx] = x;
        zout[(size_t)b * 1280 + 1024 + j] = x;
    } else {
        xh[idx] = h[b * H_ + (j - E_)];
    }
}

// gates(32x2048) -> c, h_lstm; h_lstm also into wh[:,512:1024]
__global__ void lstm_pointwise(const float* __restrict__ gates,
                               float* __restrict__ c,
                               float* __restrict__ h_lstm,
                               float* __restrict__ wh)
{
    int idx = blockIdx.x * blockDim.x + threadIdx.x;
    if (idx >= B_ * H_) return;
    int b = idx >> 9, hh = idx & (H_ - 1);
    const float* g = gates + (size_t)b * 2048;
    float ig = sigmoidf_(g[hh]);
    float fg = sigmoidf_(g[H_ + hh]);
    float gg = tanhf(g[2 * H_ + hh]);
    float og = sigmoidf_(g[3 * H_ + hh]);
    float ct = fg * c[idx] + ig * gg;
    float hl = og * tanhf(ct);
    c[idx] = ct;
    h_lstm[idx] = hl;
    wh[(size_t)b * 1024 + H_ + hh] = hl;
}

// att[b,s] = sum_h V[h] * tanh(inp[b,h] + ctx[b,s,h]); one wave per s
__global__ void att_scores(const _Float16* __restrict__ ctx,
                           const float* __restrict__ inp,
                           const float* __restrict__ V,
                           float* __restrict__ att)
{
    int b    = blockIdx.x;
    int wave = threadIdx.x >> 5;
    int lane = threadIdx.x & 31;
    int s    = blockIdx.y * 8 + wave;
    const _Float16* row = ctx + ((size_t)b * S_ + s) * H_;
    const float* inpb = inp + (size_t)b * H_;
    __builtin_prefetch(row + 8 * H_, 0, 1);       // this wave's next row
    float p = 0.0f;
#pragma unroll
    for (int j = 0; j < 16; ++j) {
        int hh = lane + 32 * j;
        p += V[hh] * tanhf(inpb[hh] + (float)row[hh]);
    }
#pragma unroll
    for (int off = 16; off; off >>= 1) p += __shfl_down(p, off, 32);
    if (lane == 0) att[(size_t)b * S_ + s] = p;
}

// softmax over s for each b (block per b)
__global__ void softmax_rows(const float* __restrict__ att, float* __restrict__ alpha)
{
    __shared__ float red[256];
    int b = blockIdx.x, tid = threadIdx.x;
    const float* a = att + (size_t)b * S_;
    float mx = -3.4e38f;
    for (int s = tid; s < S_; s += 256) mx = fmaxf(mx, a[s]);
    red[tid] = mx; __syncthreads();
    for (int off = 128; off; off >>= 1) {
        if (tid < off) red[tid] = fmaxf(red[tid], red[tid + off]);
        __syncthreads();
    }
    mx = red[0]; __syncthreads();
    float sum = 0.0f;
    for (int s = tid; s < S_; s += 256) {
        float e = __expf(a[s] - mx);
        alpha[(size_t)b * S_ + s] = e;
        sum += e;
    }
    red[tid] = sum; __syncthreads();
    for (int off = 128; off; off >>= 1) {
        if (tid < off) red[tid] += red[tid + off];
        __syncthreads();
    }
    float inv = 1.0f / red[0];
    for (int s = tid; s < S_; s += 256) alpha[(size_t)b * S_ + s] *= inv;
}

// weighted[b,h] = sum_s ctx[b,s,h]*alpha[b,s] -> wh[:,0:512] and zout[:,512:1024]
__global__ void att_weighted(const _Float16* __restrict__ ctx,
                             const float* __restrict__ alpha,
                             float* __restrict__ wh, float* __restrict__ zout)
{
    int b = blockIdx.x;
    int hh = blockIdx.y * 256 + threadIdx.x;
    const _Float16* base = ctx + (size_t)b * S_ * H_ + hh;
    const float* al = alpha + (size_t)b * S_;
    float acc = 0.0f;
    for (int s = 0; s < S_; s += 4) {
        __builtin_prefetch(base + (size_t)(s + 16) * H_, 0, 1);
        acc += al[s + 0] * (float)base[(size_t)(s + 0) * H_];
        acc += al[s + 1] * (float)base[(size_t)(s + 1) * H_];
        acc += al[s + 2] * (float)base[(size_t)(s + 2) * H_];
        acc += al[s + 3] * (float)base[(size_t)(s + 3) * H_];
    }
    wh[(size_t)b * 1024 + hh] = acc;
    zout[(size_t)b * 1280 + H_ + hh] = acc;
}

// argmax over labels, first-max tie-break; one wave per (b,t) row
__global__ void argmax_rows(const float* __restrict__ out, float* __restrict__ preds)
{
    int wid  = (int)((blockIdx.x * blockDim.x + threadIdx.x) >> 5);
    int lane = threadIdx.x & 31;
    if (wid >= B_ * T_) return;
    const float* row = out + (size_t)wid * NL_;
    float best = -3.4e38f; int bidx = 0x7fffffff;
    for (int n = lane; n < NL_; n += 32) {
        float v = row[n];
        if (v > best) { best = v; bidx = n; }
    }
#pragma unroll
    for (int off = 16; off; off >>= 1) {
        float ov = __shfl_down(best, off, 32);
        int   oi = __shfl_down(bidx, off, 32);
        if (ov > best || (ov == best && oi < bidx)) { best = ov; bidx = oi; }
    }
    if (lane == 0) preds[wid] = (float)bidx;
}

// ---------------------------------------------------------------------------
// Host orchestration
// ---------------------------------------------------------------------------
extern "C" void kernel_launch(void* const* d_in, const int* in_sizes, int n_in,
                              void* d_out, int out_size, void* d_ws, size_t ws_size,
                              hipStream_t stream)
{
    const float* dec_in    = (const float*)d_in[2];
    const float* h0        = (const float*)d_in[3];
    const float* c0        = (const float*)d_in[4];
    const float* context   = (const float*)d_in[5];
    const int*   cur_lab   = (const int*)  d_in[6];
    const float* label_emb = (const float*)d_in[7];
    const float* W_ih      = (const float*)d_in[8];
    const float* b_ih      = (const float*)d_in[9];
    const float* W_hh      = (const float*)d_in[10];
    const float* b_hh      = (const float*)d_in[11];
    const float* W_in      = (const float*)d_in[12];
    const float* b_in      = (const float*)d_in[13];
    const float* W_ctx     = (const float*)d_in[14];
    const float* b_ctx     = (const float*)d_in[15];
    const float* Vv        = (const float*)d_in[16];
    const float* W_ho      = (const float*)d_in[17];
    const float* b_ho      = (const float*)d_in[18];
    const float* W_out     = (const float*)d_in[19];
    const float* b_out     = (const float*)d_in[20];

    float* out_f = (float*)d_out;
    float* outputs = out_f;                                   // B*T*NL
    float* preds   = out_f + (size_t)B_ * T_ * NL_;           // B*T
    float* h_f     = preds + B_ * T_;                         // B*H
    float* c_f     = h_f + B_ * H_;                           // B*H

    // bump allocator over workspace
    char* ws = (char*)d_ws;
    auto alloc = [&](size_t bytes) -> void* {
        void* p = (void*)ws;
        ws += (bytes + 255) & ~(size_t)255;
        return p;
    };
    _Float16* ctx_h  = (_Float16*)alloc((size_t)B_ * S_ * H_ * 2);      // 64 MB
    _Float16* Wg_h   = (_Float16*)alloc((size_t)128 * 24 * 512 * 2);
    _Float16* Win_h  = (_Float16*)alloc((size_t)32 * 16 * 512 * 2);
    _Float16* Wctx_h = (_Float16*)alloc((size_t)32 * 16 * 512 * 2);
    _Float16* Who_h  = (_Float16*)alloc((size_t)32 * 32 * 512 * 2);
    _Float16* Wout_h = (_Float16*)alloc((size_t)128 * 40 * 512 * 2);    // N padded 2000->2048
    float* gates  = (float*)alloc((size_t)B_ * 2048 * 4);
    float* xh     = (float*)alloc((size_t)B_ * 768 * 4);
    float* hbuf   = (float*)alloc((size_t)B_ * H_ * 4);
    float* cbuf   = (float*)alloc((size_t)B_ * H_ * 4);
    float* h_lstm = (float*)alloc((size_t)B_ * H_ * 4);
    float* inp    = (float*)alloc((size_t)B_ * H_ * 4);
    float* att    = (float*)alloc((size_t)B_ * S_ * 4);
    float* alpha  = (float*)alloc((size_t)B_ * S_ * 4);
    float* wh     = (float*)alloc((size_t)B_ * 1024 * 4);
    float* zout   = (float*)alloc((size_t)B_ * 1280 * 4);

    auto blocks = [](size_t threads) { return (unsigned)((threads + 255) / 256); };

    // ---- per-call weight conversion + ctx GEMM ----------------------------
    swizzle_gates<<<blocks((size_t)128 * 24 * 512), 256, 0, stream>>>(W_ih, W_hh, Wg_h);
    swizzle_b<<<blocks((size_t)32 * 16 * 512), 256, 0, stream>>>(W_in, Win_h, H_, H_, 32, 16);
    swizzle_b<<<blocks((size_t)32 * 16 * 512), 256, 0, stream>>>(W_ctx, Wctx_h, H_, H_, 32, 16);
    swizzle_b<<<blocks((size_t)32 * 32 * 512), 256, 0, stream>>>(W_ho, Who_h, H_, 1024, 32, 32);
    swizzle_b<<<blocks((size_t)128 * 40 * 512), 256, 0, stream>>>(W_out, Wout_h, NL_, 1280, 128, 40);

    // ctx: 4096 Mtiles x 8 Ngroups waves = 32768 waves
    gemm_wmma_ctx4<<<(4096u * 8u * 32u) / 256u, 256, 0, stream>>>(context, Wctx_h, b_ctx, ctx_h);

    // init recurrent state
    hipMemcpyAsync(hbuf, h0, (size_t)B_ * H_ * 4, hipMemcpyDeviceToDevice, stream);
    hipMemcpyAsync(cbuf, c0, (size_t)B_ * H_ * 4, hipMemcpyDeviceToDevice, stream);

    // ---- recurrence -------------------------------------------------------
    for (int t = 0; t < T_; ++t) {
        build_x<<<blocks(B_ * 768), 256, 0, stream>>>(dec_in, cur_lab, label_emb,
                                                      hbuf, xh, zout, t);
        // gates = xh @ [W_ih|W_hh]^T + b_ih + b_hh : Mtiles=2, Ngroups=32, Ktiles=24
        gemm_wmma4<<<blocks((size_t)2 * 32 * 32), 256, 0, stream>>>(
            xh, 768, Wg_h, 24, b_ih, b_hh, gates, 2048, nullptr, 0, 2, 32, 2048, 0);
        lstm_pointwise<<<blocks(B_ * H_), 256, 0, stream>>>(gates, cbuf, h_lstm, wh);
        // inp = h_lstm @ W_in^T + b_in : Mtiles=2, Ngroups=8, Ktiles=16
        gemm_wmma4<<<blocks((size_t)2 * 8 * 32), 256, 0, stream>>>(
            h_lstm, H_, Win_h, 16, b_in, nullptr, inp, H_, nullptr, 0, 2, 8, H_, 0);
        att_scores<<<dim3(B_, S_ / 8), 256, 0, stream>>>(ctx_h, inp, Vv, att);
        softmax_rows<<<B_, 256, 0, stream>>>(att, alpha);
        att_weighted<<<dim3(B_, H_ / 256), 256, 0, stream>>>(ctx_h, alpha, wh, zout);
        // h_t = tanh(wh @ W_ho^T + b_ho) -> hbuf and zout[:,0:512]
        gemm_wmma4<<<blocks((size_t)2 * 8 * 32), 256, 0, stream>>>(
            wh, 1024, Who_h, 32, b_ho, nullptr, hbuf, H_, zout, 1280, 2, 8, H_, 1);
        // out = zout @ W_out^T + b_out -> outputs[:, t, :]
        gemm_wmma4<<<blocks((size_t)2 * 32 * 32), 256, 0, stream>>>(
            zout, 1280, Wout_h, 40, b_out, nullptr,
            outputs + (size_t)t * NL_, T_ * NL_, nullptr, 0, 2, 32, NL_, 0);
    }

    // ---- epilogue ---------------------------------------------------------
    argmax_rows<<<blocks((size_t)B_ * T_ * 32), 256, 0, stream>>>(outputs, preds);
    hipMemcpyAsync(h_f, hbuf, (size_t)B_ * H_ * 4, hipMemcpyDeviceToDevice, stream);
    hipMemcpyAsync(c_f, cbuf, (size_t)B_ * H_ * 4, hipMemcpyDeviceToDevice, stream);

    (void)in_sizes; (void)n_in; (void)out_size; (void)ws_size; (void)d_in;
}
